// MambaModel_13838384628223
// MI455X (gfx1250) — matrix operand order repaired and checked by reading, not compiled, not run on hardware
//
#include <hip/hip_runtime.h>
#include <hip/hip_bf16.h>

// ---------------- constants ----------------
#define B_SZ     64
#define SEQ      2048
#define MROWS    (B_SZ * SEQ)     // 131072
#define IN_DIM   16
#define D_MODEL  128
#define D_INNER  256
#define D_STATE  16
#define N_LAYERS 5

// ---------------- WMMA vector types ----------------
typedef __attribute__((ext_vector_type(16))) __bf16 bf16x16;
typedef __attribute__((ext_vector_type(8)))  float  floatx8;
typedef __attribute__((ext_vector_type(4)))  int    v4i;

#if defined(__has_builtin)
#if __has_builtin(__builtin_amdgcn_global_load_async_to_lds_b128)
#define HAVE_ASYNC_LDS 1
#endif
#endif

// fp32 -> bf16 (round to nearest even), stored as raw u16
__device__ __forceinline__ unsigned short f2bf(float f) {
    unsigned int u = __float_as_uint(f);
    unsigned int r = u + 0x7FFFu + ((u >> 16) & 1u);
    return (unsigned short)(r >> 16);
}

__device__ __forceinline__ float sigmoidf(float x) { return 1.0f / (1.0f + __expf(-x)); }
__device__ __forceinline__ float siluf(float x)    { return x * sigmoidf(x); }
__device__ __forceinline__ float softplusf(float x){ return (x > 20.0f) ? x : log1pf(__expf(x)); }

// epilogue modes
#define MODE_NONE       0
#define MODE_SOFTPLUS   2
#define MODE_SPLIT_SILU 3   // silu on cols n < split, identity otherwise

// ---------------- weight transpose + bf16 convert: dst[N x K] = bf16(src[K x N]) ----------------
__global__ void wconv_kernel(const float* __restrict__ src, unsigned short* __restrict__ dst,
                             int K, int N) {
    int idx = blockIdx.x * blockDim.x + threadIdx.x;
    if (idx >= K * N) return;
    int k = idx / N, n = idx % N;
    dst[(size_t)n * K + k] = f2bf(src[idx]);
}

// ---------------- embedding: h = x @ emb_w + emb_b (K=16, plain VALU) ----------------
__global__ void embed_kernel(const float* __restrict__ x, const float* __restrict__ w,
                             const float* __restrict__ b,
                             float* __restrict__ hf, unsigned short* __restrict__ hb) {
    long long idx = (long long)blockIdx.x * blockDim.x + threadIdx.x;
    if (idx >= (long long)MROWS * D_MODEL) return;
    long long row = idx >> 7;
    int col = (int)(idx & 127);
    const float* xr = x + row * IN_DIM;
    float acc = b[col];
#pragma unroll
    for (int k = 0; k < IN_DIM; ++k) acc += xr[k] * w[k * D_MODEL + col];
    hf[idx] = acc;
    hb[idx] = f2bf(acc);
}

// ---------------- bf16 WMMA GEMM with LDS-staged B panel ----------------
// Block = 4 waves; all waves share one (NT*16)-column B panel (async-staged to LDS),
// each wave owns a distinct 32-row M tile -> 128 rows x NT*16 cols per block.
// K is compile-time, so the k-loop fully unrolls into global A loads + ds_load B + v_wmma.
template <int NT, int KT, int MODE, bool WF, bool WH>
__global__ void gemm_bf16_wmma(const unsigned short* __restrict__ A, int lda,
                               const unsigned short* __restrict__ W,   // [N x K] bf16
                               const float* __restrict__ bias,
                               int M, int N,
                               float* __restrict__ outF, int ldoF,
                               unsigned short* __restrict__ outH, int ldoH,
                               int split) {
    constexpr int PROWS = NT * 16;       // B-panel rows (columns of output)
    constexpr int LDK   = KT + 8;        // LDS row stride (ushorts); +16B pad -> bank-spread
    __shared__ unsigned short sB[PROWS * LDK];

    const int tid  = threadIdx.x;
    const int lane = tid & 31;
    const int wave = tid >> 5;
    const int l15  = lane & 15;
    const int half = lane >> 4;

    const int ntb     = N / (NT * 16);
    const int nb      = blockIdx.x % ntb;
    const int mtgroup = blockIdx.x / ntb;
    const int n0base  = nb * (NT * 16);
    const int m0      = (mtgroup * 4 + wave) << 5;   // 32 rows per wave

    // ---- stage B panel [PROWS x KT] into LDS (shared by all 4 waves) ----
    {
        constexpr int NCHUNK = PROWS * (KT / 8);     // 16B chunks
#pragma unroll
        for (int c = tid; c < NCHUNK; c += 128) {
            const int row  = c / (KT / 8);
            const int colc = c - row * (KT / 8);
            const unsigned short* gsrc = W + (size_t)(n0base + row) * KT + colc * 8;
            const int ldst = row * LDK + colc * 8;
#ifdef HAVE_ASYNC_LDS
            __builtin_amdgcn_global_load_async_to_lds_b128(
                (v4i*)gsrc, (v4i*)&sB[ldst], 0, 0);
#else
            *(uint4*)&sB[ldst] = *(const uint4*)gsrc;
#endif
        }
#ifdef HAVE_ASYNC_LDS
#if __has_builtin(__builtin_amdgcn_s_wait_asynccnt)
        __builtin_amdgcn_s_wait_asynccnt(0);
#else
        asm volatile("s_wait_asynccnt 0x0" ::: "memory");
#endif
#endif
        __syncthreads();
    }

    // init accumulators with bias (bias depends only on column)
    floatx8 acc[2][NT];
#pragma unroll
    for (int j = 0; j < NT; ++j) {
        const float bv = bias[n0base + j * 16 + l15];
#pragma unroll
        for (int r = 0; r < 8; ++r) { acc[0][j][r] = bv; acc[1][j][r] = bv; }
    }

    // A fragment rows (16-bit A layout: lanes 0-15 hold K 0-7 & 16-23; lanes 16-31 hold K 8-15 & 24-31)
    const unsigned short* Arow0 = A + (size_t)(m0 + l15) * lda;
    const unsigned short* Arow1 = Arow0 + (size_t)16 * lda;

#pragma unroll
    for (int k0 = 0; k0 < KT; k0 += 32) {
        union { bf16x16 v; uint4 q[2]; } a0, a1;
        a0.q[0] = *(const uint4*)(Arow0 + k0 + half * 8);
        a0.q[1] = *(const uint4*)(Arow0 + k0 + 16 + half * 8);
        a1.q[0] = *(const uint4*)(Arow1 + k0 + half * 8);
        a1.q[1] = *(const uint4*)(Arow1 + k0 + 16 + half * 8);

#pragma unroll
        for (int j = 0; j < NT; ++j) {
            // B fragment from LDS: lanes 0-15 hold K 0-7 & 16-23, lanes 16-31 hold K 8-15 & 24-31
            const unsigned short* sRow = &sB[(j * 16 + l15) * LDK + k0 + half * 8];
            union { bf16x16 v; uint4 q[2]; } bg;
            bg.q[0] = *(const uint4*)(sRow);
            bg.q[1] = *(const uint4*)(sRow + 16);
            acc[0][j] = __builtin_amdgcn_wmma_f32_16x16x32_bf16(
                            false, a0.v, false, bg.v, (short)0, acc[0][j], false, false);
            acc[1][j] = __builtin_amdgcn_wmma_f32_16x16x32_bf16(
                            false, a1.v, false, bg.v, (short)0, acc[1][j], false, false);
        }
    }

    // epilogue: activation + stores (C layout: lane -> col, VGPR r -> row m0 + r + 8*half)
#pragma unroll
    for (int i = 0; i < 2; ++i) {
#pragma unroll
        for (int j = 0; j < NT; ++j) {
            const int n = n0base + j * 16 + l15;
#pragma unroll
            for (int r = 0; r < 8; ++r) {
                float v = acc[i][j][r];
                if constexpr (MODE == MODE_SOFTPLUS)   v = softplusf(v);
                if constexpr (MODE == MODE_SPLIT_SILU) { if (n < split) v = siluf(v); }
                const size_t mrow = (size_t)(m0 + i * 16 + r + half * 8);
                if constexpr (WF) outF[mrow * ldoF + n] = v;
                if constexpr (WH) outH[mrow * ldoH + n] = f2bf(v);
            }
        }
    }
}

// ---------------- selective-scan: one block per batch, one thread per d_inner channel ----------------
__global__ void scan_kernel(const float* __restrict__ xz,   // [MROWS x 512]; cols 0..255 = u
                            const float* __restrict__ dt,   // [MROWS x 256] softplus'd
                            const float* __restrict__ bc,   // [MROWS x 32]  (B | C)
                            const float* __restrict__ alog, // [256 x 16] this layer
                            const float* __restrict__ dp,   // [256] this layer
                            float* __restrict__ y) {        // [MROWS x 256]
    const int b = blockIdx.x;
    const int d = threadIdx.x;

    float Aa[D_STATE];
#pragma unroll
    for (int n = 0; n < D_STATE; ++n) Aa[n] = -__expf(alog[d * D_STATE + n]);
    const float Dv = dp[d];

    float h[D_STATE];
#pragma unroll
    for (int n = 0; n < D_STATE; ++n) h[n] = 0.0f;

    __shared__ float sbc[32];

    for (int t = 0; t < SEQ; ++t) {
        const size_t row = (size_t)b * SEQ + t;
        if (d < 32) sbc[d] = bc[row * 32 + d];
        __syncthreads();

        const float u   = xz[row * 512 + d];
        const float dtv = dt[row * 256 + d];
        const float du  = dtv * u;
        float acc = Dv * u;
#pragma unroll
        for (int n = 0; n < D_STATE; ++n) {
            const float dA = __expf(dtv * Aa[n]);
            h[n] = h[n] * dA + du * sbc[n];
            acc += h[n] * sbc[16 + n];
        }
        y[row * 256 + d] = acc;
        __syncthreads();
    }
}

// ---------------- gating: yg = bf16( y * silu(z) ) ----------------
__global__ void gate_kernel(const float* __restrict__ y, const float* __restrict__ xz,
                            unsigned short* __restrict__ yg) {
    long long idx = (long long)blockIdx.x * blockDim.x + threadIdx.x;
    if (idx >= (long long)MROWS * D_INNER) return;
    long long row = idx >> 8;
    int col = (int)(idx & 255);
    const float z = xz[row * 512 + 256 + col];
    yg[idx] = f2bf(y[idx] * siluf(z));
}

// ---------------- residual add + LayerNorm (one block per row of 128) ----------------
__global__ void addnorm_kernel(const float* __restrict__ ho,
                               const float* __restrict__ g, const float* __restrict__ be,
                               float* __restrict__ hf, unsigned short* __restrict__ hb) {
    const size_t row = blockIdx.x;
    const int t = threadIdx.x;
    const float v = ho[row * D_MODEL + t] + hf[row * D_MODEL + t];

    __shared__ float red[D_MODEL];
    red[t] = v; __syncthreads();
    for (int s = 64; s > 0; s >>= 1) { if (t < s) red[t] += red[t + s]; __syncthreads(); }
    const float mu = red[0] * (1.0f / D_MODEL);
    __syncthreads();
    const float dv = v - mu;
    red[t] = dv * dv; __syncthreads();
    for (int s = 64; s > 0; s >>= 1) { if (t < s) red[t] += red[t + s]; __syncthreads(); }
    const float var = red[0] * (1.0f / D_MODEL);

    const float o = dv * rsqrtf(var + 1e-5f) * g[t] + be[t];
    hf[row * D_MODEL + t] = o;
    hb[row * D_MODEL + t] = f2bf(o);
}

// ---------------- mean pool over sequence ----------------
__global__ void pool_kernel(const float* __restrict__ hf, float* __restrict__ pooled) {
    const int b = blockIdx.x;
    const int t = threadIdx.x;
    float s = 0.0f;
    for (int q = 0; q < SEQ; ++q) s += hf[((size_t)b * SEQ + q) * D_MODEL + t];
    pooled[b * D_MODEL + t] = s * (1.0f / SEQ);
}

// ---------------- head MLP + final pricing formula ----------------
__global__ void head_kernel(const float* __restrict__ pooled,
                            const float* __restrict__ o1w, const float* __restrict__ o1b,
                            const float* __restrict__ o2w, const float* __restrict__ o2b,
                            const float* __restrict__ origins,
                            float* __restrict__ out) {
    const int b = threadIdx.x;
    if (b >= B_SZ) return;
    const float* p = pooled + b * D_MODEL;

    float hid[64];
    for (int j = 0; j < 64; ++j) {
        float acc = o1b[j];
        for (int k = 0; k < D_MODEL; ++k) acc += p[k] * o1w[k * 64 + j];
        hid[j] = fmaxf(acc, 0.0f);
    }
    float prm[3];
    for (int j = 0; j < 3; ++j) {
        float acc = o2b[j];
        for (int k = 0; k < 64; ++k) acc += hid[k] * o2w[k * 3 + j];
        prm[j] = 2.0f * sigmoidf(acc);
    }
    const float* L1 = origins + ((size_t)b * SEQ + (SEQ - 1)) * IN_DIM;
    const float* L2 = origins + ((size_t)b * SEQ + (SEQ - 2)) * IN_DIM;
    const float price = L1[0];
    const float ry_n = L1[1], ry_b = L2[1];
    const float gy_n = L1[2], gy_b = L2[2];
    const float ny_n = L1[3], ny_b = L2[3];
    out[b] = price * prm[0] * ry_n / ry_b * 2.0f *
             (prm[1] * sigmoidf(gy_n - gy_b) + prm[2] * sigmoidf(ny_n - ny_b));
}

// ---------------- host orchestration ----------------
extern "C" void kernel_launch(void* const* d_in, const int* in_sizes, int n_in,
                              void* d_out, int out_size, void* d_ws, size_t ws_size,
                              hipStream_t stream) {
    (void)in_sizes; (void)n_in; (void)out_size; (void)ws_size;
    const float* origins = (const float*)d_in[0];
    const float* x       = (const float*)d_in[1];
    const float* emb_w   = (const float*)d_in[2];
    const float* emb_b   = (const float*)d_in[3];
    const float* in_w    = (const float*)d_in[4];
    const float* in_b    = (const float*)d_in[5];
    const float* xp_w    = (const float*)d_in[6];
    const float* xp_b    = (const float*)d_in[7];
    const float* dt_w    = (const float*)d_in[8];
    const float* dt_b    = (const float*)d_in[9];
    const float* out_w   = (const float*)d_in[10];
    const float* out_b   = (const float*)d_in[11];
    const float* A_log   = (const float*)d_in[12];
    const float* Dp      = (const float*)d_in[13];
    const float* ln_g    = (const float*)d_in[14];
    const float* ln_b    = (const float*)d_in[15];
    const float* o1_w    = (const float*)d_in[16];
    const float* o1_b    = (const float*)d_in[17];
    const float* o2_w    = (const float*)d_in[18];
    const float* o2_b    = (const float*)d_in[19];
    float* out = (float*)d_out;

    // workspace carve-out
    char* base = (char*)d_ws;
    size_t off = 0;
    auto carve = [&](size_t bytes) -> char* {
        char* p = base + off;
        off += (bytes + 255) & ~(size_t)255;
        return p;
    };
    const size_t M = MROWS;
    float*          h_f    = (float*)         carve(M * 128 * 4);
    unsigned short* h_bf   = (unsigned short*)carve(M * 128 * 2);
    float*          xz_f   = (float*)         carve(M * 512 * 4);
    unsigned short* xz_h   = (unsigned short*)carve(M * 512 * 2);
    float*          bc_f   = (float*)         carve(M * 32 * 4);
    float*          dt_f   = (float*)         carve(M * 256 * 4);
    float*          y_f    = (float*)         carve(M * 256 * 4);
    unsigned short* yg_h   = (unsigned short*)carve(M * 256 * 2);
    float*          ho_f   = (float*)         carve(M * 128 * 4);
    float*          pooled = (float*)         carve(64 * 128 * 4);
    unsigned short* w_in_t = (unsigned short*)carve((size_t)N_LAYERS * 512 * 128 * 2);
    unsigned short* w_xp_t = (unsigned short*)carve((size_t)N_LAYERS * 32 * 256 * 2);
    unsigned short* w_dt_t = (unsigned short*)carve((size_t)N_LAYERS * 256 * 256 * 2);
    unsigned short* w_ot_t = (unsigned short*)carve((size_t)N_LAYERS * 128 * 256 * 2);

    // 1) transpose+bf16-convert weights
    for (int l = 0; l < N_LAYERS; ++l) {
        wconv_kernel<<<(128 * 512 + 255) / 256, 256, 0, stream>>>(
            in_w + (size_t)l * 128 * 512, w_in_t + (size_t)l * 512 * 128, 128, 512);
        wconv_kernel<<<(256 * 32 + 255) / 256, 256, 0, stream>>>(
            xp_w + (size_t)l * 256 * 32, w_xp_t + (size_t)l * 32 * 256, 256, 32);
        wconv_kernel<<<(256 * 256 + 255) / 256, 256, 0, stream>>>(
            dt_w + (size_t)l * 256 * 256, w_dt_t + (size_t)l * 256 * 256, 256, 256);
        wconv_kernel<<<(256 * 128 + 255) / 256, 256, 0, stream>>>(
            out_w + (size_t)l * 256 * 128, w_ot_t + (size_t)l * 128 * 256, 256, 128);
    }

    // 2) embedding
    {
        long long tot = (long long)M * 128;
        embed_kernel<<<(unsigned)((tot + 255) / 256), 256, 0, stream>>>(x, emb_w, emb_b, h_f, h_bf);
    }

    // blocks: 4 waves x 32 rows each, sharing one NT*16-col B panel
    auto gemm_blocks = [&](int Mm, int Nn, int nt) -> unsigned {
        return (unsigned)((Mm >> 7) * (Nn / (nt * 16)));
    };

    // 3) layers
    for (int l = 0; l < N_LAYERS; ++l) {
        // in_proj: h[M,128] -> xz[M,512]; silu on first 256 cols; fp32 + bf16 outputs
        gemm_bf16_wmma<4, 128, MODE_SPLIT_SILU, true, true><<<gemm_blocks(M, 512, 4), 128, 0, stream>>>(
            h_bf, 128, w_in_t + (size_t)l * 512 * 128, in_b + (size_t)l * 512,
            M, 512, xz_f, 512, xz_h, 512, 256);
        // x_proj: u[M,256] -> (B|C)[M,32]
        gemm_bf16_wmma<2, 256, MODE_NONE, true, false><<<gemm_blocks(M, 32, 2), 128, 0, stream>>>(
            xz_h, 512, w_xp_t + (size_t)l * 32 * 256, xp_b + (size_t)l * 32,
            M, 32, bc_f, 32, nullptr, 0, 0);
        // dt_proj: u[M,256] -> softplus -> dt[M,256]
        gemm_bf16_wmma<4, 256, MODE_SOFTPLUS, true, false><<<gemm_blocks(M, 256, 4), 128, 0, stream>>>(
            xz_h, 512, w_dt_t + (size_t)l * 256 * 256, dt_b + (size_t)l * 256,
            M, 256, dt_f, 256, nullptr, 0, 0);
        // selective scan
        scan_kernel<<<B_SZ, 256, 0, stream>>>(
            xz_f, dt_f, bc_f, A_log + (size_t)l * 256 * 16, Dp + (size_t)l * 256, y_f);
        // gating
        {
            long long tot = (long long)M * 256;
            gate_kernel<<<(unsigned)((tot + 255) / 256), 256, 0, stream>>>(y_f, xz_f, yg_h);
        }
        // out_proj: yg[M,256] -> ho[M,128]
        gemm_bf16_wmma<4, 256, MODE_NONE, true, false><<<gemm_blocks(M, 128, 4), 128, 0, stream>>>(
            yg_h, 256, w_ot_t + (size_t)l * 128 * 256, out_b + (size_t)l * 128,
            M, 128, ho_f, 128, nullptr, 0, 0);
        // residual + layernorm (updates h_f / h_bf in place)
        addnorm_kernel<<<(unsigned)M, 128, 0, stream>>>(
            ho_f, ln_g + (size_t)l * 128, ln_b + (size_t)l * 128, h_f, h_bf);
    }

    // 4) pool + head
    pool_kernel<<<B_SZ, 128, 0, stream>>>(h_f, pooled);
    head_kernel<<<1, 64, 0, stream>>>(pooled, o1_w, o1_b, o2_w, o2_b, origins, out);
}
